// GCN_55499567399154
// MI455X (gfx1250) — matrix operand order repaired and checked by reading, not compiled, not run on hardware
//
#include <hip/hip_runtime.h>

typedef __bf16 bf16;
typedef __attribute__((ext_vector_type(16))) __bf16 bf16x16;
typedef __attribute__((ext_vector_type(8)))  __bf16 bf16x8;
typedef __attribute__((ext_vector_type(8)))  float   v8f;

#define F_IN 128

// ---------------- utility kernels ----------------
__global__ void k_zero(float* __restrict__ p, int n) {
  int i = blockIdx.x * blockDim.x + threadIdx.x;
  if (i < n) p[i] = 0.0f;
}

__global__ void k_deg(const int* __restrict__ col, float* __restrict__ deg, int E) {
  int e = blockIdx.x * blockDim.x + threadIdx.x;
  if (e < E) atomicAdd(&deg[col[e]], 1.0f);
}

// dinv[i] = rsqrt(deg[i] + 1)   (+1 = self loop), in-place over deg buffer
__global__ void k_dinv(float* __restrict__ d, int n) {
  int i = blockIdx.x * blockDim.x + threadIdx.x;
  if (i < n) d[i] = rsqrtf(d[i] + 1.0f);
}

__global__ void k_cvt_bf16(const float* __restrict__ x, bf16* __restrict__ y, int n) {
  int i = blockIdx.x * blockDim.x + threadIdx.x;
  if (i < n) y[i] = (bf16)x[i];
}

// ---------------- WMMA GEMM:  H[N x OUT] = Xb[N x 128] @ W[128 x OUT] ----------------
// One wave per 16-row M tile. K=128 in 4 steps of K=32 (v_wmma_f32_16x16x32_bf16).
// W is transposed into LDS as bf16 with padded stride so B loads are contiguous.
template <int OUT>
__global__ __launch_bounds__(128) void k_gemm(const bf16* __restrict__ Xb,
                                              const float* __restrict__ W,   // [128][OUT] fp32
                                              float* __restrict__ H,         // [N][OUT]
                                              int nrows, int mtiles) {
  constexpr int LDW = 136;                 // padded K-stride (elements); 136*2B = 272B = 16B-aligned
  __shared__ bf16 Wt[OUT * LDW];           // transposed weights: Wt[n][k]
  const int tid = threadIdx.x;

  for (int idx = tid; idx < 128 * OUT; idx += 128) {
    int k = idx / OUT;
    int n = idx % OUT;
    Wt[n * LDW + k] = (bf16)W[idx];
  }
  __syncthreads();

  const int wave  = tid >> 5;
  const int lane  = tid & 31;
  const int mt    = blockIdx.x * 4 + wave;
  if (mt >= mtiles) return;
  const int m0    = mt * 16;
  const int lhalf = lane >> 4;             // 0: lanes 0-15, 1: lanes 16-31
  const int lrow  = lane & 15;

  int mrow = m0 + lrow;                    // A source row for this lane
  if (mrow >= nrows) mrow = nrows - 1;     // clamp (stores are guarded)
  const bf16* arow = Xb + (size_t)mrow * F_IN;

  constexpr int NT = OUT / 16;
  v8f acc[NT] = {};

  #pragma unroll
  for (int kk = 0; kk < 4; ++kk) {         // K step = 32
    union { bf16x16 v; bf16x8 h[2]; } a;
    const int c0 = kk * 4 + lhalf;         // 8-element chunk index
    a.h[0] = *(const bf16x8*)(arow + c0 * 8);        // K = kb + lhalf*8 .. +7
    a.h[1] = *(const bf16x8*)(arow + (c0 + 2) * 8);  // K = kb + lhalf*8 + 16 .. +23
    #pragma unroll
    for (int nt = 0; nt < NT; ++nt) {
      union { bf16x16 v; bf16x8 h[2]; } b;
      const bf16* bp = &Wt[(nt * 16 + lrow) * LDW + kk * 32 + lhalf * 8];
      b.h[0] = *(const bf16x8*)(bp);
      b.h[1] = *(const bf16x8*)(bp + 16);
      acc[nt] = __builtin_amdgcn_wmma_f32_16x16x32_bf16(
          /*neg_a=*/false, a.v, /*neg_b=*/false, b.v,
          /*c_mod=*/(short)0, acc[nt], /*reuse_a=*/false, /*reuse_b=*/false);
    }
  }

  // C/D layout: VGPR r, lane l -> M = r + 8*lhalf, N = lrow
  #pragma unroll
  for (int nt = 0; nt < NT; ++nt) {
    #pragma unroll
    for (int r = 0; r < 8; ++r) {
      int m = m0 + r + 8 * lhalf;
      if (m < nrows) H[(size_t)m * OUT + nt * 16 + lrow] = acc[nt][r];
    }
  }
}

// ---------------- edge scatter:  acc[col] += H[row] * dinv[row]*dinv[col] ----------------
// One wave per edge; each lane owns F/32 consecutive features.
template <int F>
__global__ __launch_bounds__(256) void k_scatter(const float* __restrict__ H,
                                                 const int* __restrict__ row,
                                                 const int* __restrict__ col,
                                                 const float* __restrict__ dinv,
                                                 float* __restrict__ acc, int E) {
  const int w    = (blockIdx.x * blockDim.x + threadIdx.x) >> 5;
  const int lane = threadIdx.x & 31;
  if (w >= E) return;
  const int r = row[w];
  const int c = col[w];
  const float nrm = dinv[r] * dinv[c];
  constexpr int CH = F / 32;               // 4 (F=128) or 2 (F=64)
  const float* src = H   + (size_t)r * F + lane * CH;
  float*       dst = acc + (size_t)c * F + lane * CH;
  if constexpr (CH == 4) {
    float4 v = *(const float4*)src;
    atomicAdd(dst + 0, v.x * nrm);
    atomicAdd(dst + 1, v.y * nrm);
    atomicAdd(dst + 2, v.z * nrm);
    atomicAdd(dst + 3, v.w * nrm);
  } else {
    float2 v = *(const float2*)src;
    atomicAdd(dst + 0, v.x * nrm);
    atomicAdd(dst + 1, v.y * nrm);
  }
}

// ---------------- epilogue: self-loop + bias + (ReLU) ; emit next bf16 input / final f32 ----------------
template <int F>
__global__ void k_epilogue(const float* __restrict__ acc, const float* __restrict__ H,
                           const float* __restrict__ dinv, const float* __restrict__ bias,
                           bf16* __restrict__ xnext, float* __restrict__ outf,
                           int n, int relu) {
  int idx = blockIdx.x * blockDim.x + threadIdx.x;
  if (idx >= n * F) return;
  int i = idx / F;
  int f = idx % F;
  float dv = dinv[i];
  float v  = acc[idx] + H[idx] * dv * dv + bias[f];
  if (relu) v = fmaxf(v, 0.0f);
  if (xnext) xnext[idx] = (bf16)v;
  if (outf)  outf[idx]  = v;
}

// ---------------- host side ----------------
static inline size_t align256(size_t x) { return (x + 255) & ~(size_t)255; }

extern "C" void kernel_launch(void* const* d_in, const int* in_sizes, int n_in,
                              void* d_out, int out_size, void* d_ws, size_t ws_size,
                              hipStream_t stream) {
  const float* x  = (const float*)d_in[0];
  const int*   ei = (const int*)d_in[1];         // edge_index flat (2,E): row then col
  const float* W1 = (const float*)d_in[2];  const float* b1 = (const float*)d_in[3];
  const float* W2 = (const float*)d_in[4];  const float* b2 = (const float*)d_in[5];
  const float* W3 = (const float*)d_in[6];  const float* b3 = (const float*)d_in[7];
  const float* W4 = (const float*)d_in[8];  const float* b4 = (const float*)d_in[9];

  const int N    = in_sizes[0] / F_IN;
  const int E    = in_sizes[1] / 2;
  const int NCLS = in_sizes[9];                  // 64
  const int* row = ei;
  const int* col = ei + E;

  // workspace layout
  char* ws = (char*)d_ws;
  float* dinv = (float*)ws;                ws += align256((size_t)N * 4);
  bf16*  Xb   = (bf16*)ws;                 ws += align256((size_t)N * F_IN * 2);
  float* H    = (float*)ws;                ws += align256((size_t)N * F_IN * 4);
  float* ACC  = (float*)ws;                ws += align256((size_t)N * F_IN * 4);
  (void)ws_size;

  const int T = 256;
  auto cdiv = [](int a, int b) { return (a + b - 1) / b; };

  // graph normalization: dinv = rsqrt(in-degree + 1)
  k_zero<<<cdiv(N, T), T, 0, stream>>>(dinv, N);
  k_deg <<<cdiv(E, T), T, 0, stream>>>(col, dinv, E);
  k_dinv<<<cdiv(N, T), T, 0, stream>>>(dinv, N);

  // bf16 activations for layer 1
  k_cvt_bf16<<<cdiv(N * F_IN, T), T, 0, stream>>>(x, Xb, N * F_IN);

  const int mtiles = (N + 15) / 16;
  const int gblk   = (mtiles + 3) / 4;           // 4 waves (M tiles) per block
  const int sblk   = cdiv(E * 32, T);            // one wave per edge

  // layers 1..3: OUT=128, ReLU, emit bf16 next input
  const float* Ws[3] = {W1, W2, W3};
  const float* bs[3] = {b1, b2, b3};
  for (int l = 0; l < 3; ++l) {
    k_gemm<128><<<gblk, 128, 0, stream>>>(Xb, Ws[l], H, N, mtiles);
    k_zero<<<cdiv(N * 128, T), T, 0, stream>>>(ACC, N * 128);
    k_scatter<128><<<sblk, T, 0, stream>>>(H, row, col, dinv, ACC, E);
    k_epilogue<128><<<cdiv(N * 128, T), T, 0, stream>>>(ACC, H, dinv, bs[l],
                                                        Xb, nullptr, N, 1);
  }

  // layer 4: OUT=64, no ReLU, write f32 result
  (void)NCLS;
  k_gemm<64><<<gblk, 128, 0, stream>>>(Xb, W4, H, N, mtiles);
  k_zero<<<cdiv(N * 64, T), T, 0, stream>>>(ACC, N * 64);
  k_scatter<64><<<sblk, T, 0, stream>>>(H, row, col, dinv, ACC, E);
  k_epilogue<64><<<cdiv(N * 64, T), T, 0, stream>>>(ACC, H, dinv, b4,
                                                    nullptr, (float*)d_out, N, 0);
  (void)out_size;
}
